// RatMultiHeadedAttention_36361193128443
// MI455X (gfx1250) — compile-verified, matrix operand order checked by reading, not run on hardware
//
#include <hip/hip_runtime.h>
#include <hip/hip_bf16.h>

// RAT multi-head attention for gfx1250 (MI455X).
// Memory-bound (~0.8 GB traffic, ~25 GFLOP) -> v_wmma_f32_16x16x32_bf16 for all
// matrix math. Weights/activations/probs pre-converted once to bf16 (streaming),
// V stored transposed so every GEMM B-operand is contiguous b128 loads.
// K-loops are software-pipelined with a peeled epilogue (no in-loop branch).

#define B_   4
#define S_   512
#define HID_ 1024
#define NH_  16
#define DH_  64

#ifndef __has_builtin
#define __has_builtin(x) 0
#endif

typedef __attribute__((ext_vector_type(16))) __bf16 v16bf;
typedef __attribute__((ext_vector_type(2)))  __bf16 v2bf;
typedef __attribute__((ext_vector_type(8)))  float  v8f;

static __device__ __forceinline__ __bf16 f2bf(float f) {
  return (__bf16)f;
}

static __device__ __forceinline__ float bf2f(__bf16 b) {
  union { unsigned short s; __bf16 b; } i; i.b = b;
  union { unsigned u; float f; } o; o.u = ((unsigned)i.s) << 16;
  return o.f;
}

static __device__ __forceinline__ v2bf pk_bf16(float x, float y) {
#if __has_builtin(__builtin_amdgcn_cvt_pk_bf16_f32)
  return __builtin_amdgcn_cvt_pk_bf16_f32(x, y);
#else
  v2bf r; r[0] = f2bf(x); r[1] = f2bf(y);
  return r;
#endif
}

static __device__ __forceinline__ v8f wmma_bf16(v16bf a, v16bf b, v8f c) {
  return __builtin_amdgcn_wmma_f32_16x16x32_bf16(false, a, false, b, (short)0, c,
                                                 false, false);
}

// Raw 16-float fragment payload (two contiguous K-groups of 8).
struct f4x4 { float4 x0, x1, x2, x3; };

static __device__ __forceinline__ v16bf cvt16(const f4x4& r) {
  v16bf f; v2bf p;
  p = pk_bf16(r.x0.x, r.x0.y); f[0]  = p[0]; f[1]  = p[1];
  p = pk_bf16(r.x0.z, r.x0.w); f[2]  = p[0]; f[3]  = p[1];
  p = pk_bf16(r.x1.x, r.x1.y); f[4]  = p[0]; f[5]  = p[1];
  p = pk_bf16(r.x1.z, r.x1.w); f[6]  = p[0]; f[7]  = p[1];
  p = pk_bf16(r.x2.x, r.x2.y); f[8]  = p[0]; f[9]  = p[1];
  p = pk_bf16(r.x2.z, r.x2.w); f[10] = p[0]; f[11] = p[1];
  p = pk_bf16(r.x3.x, r.x3.y); f[12] = p[0]; f[13] = p[1];
  p = pk_bf16(r.x3.z, r.x3.w); f[14] = p[0]; f[15] = p[1];
  return f;
}

// ---- fragment loaders -----------------------------------------------------
// ISA 16-bit A/B lane layout: lane<16 -> row/col = lane, K = {k0..+7, k0+16..+23};
//                             lane>=16 -> row/col = lane-16, K groups shifted +8.

// f32 source, K contiguous (B: element(k,n) at base + n*nstride + k): raw fetch.
static __device__ __forceinline__ f4x4
fetch_bfrag_f32_kcontig(const float* __restrict__ base, long nstride, int k0, int n0) {
  const int lane = threadIdx.x & 31;
  const int l16 = lane & 15, hi = lane >> 4;
  const float* colp = base + (long)(n0 + l16) * nstride;
  const int g0 = k0 + hi * 8, g1 = g0 + 16;
  f4x4 r;
  r.x0 = *(const float4*)(colp + g0);
  r.x1 = *(const float4*)(colp + g0 + 4);
  r.x2 = *(const float4*)(colp + g1);
  r.x3 = *(const float4*)(colp + g1 + 4);
  return r;
}

// f32 source, K strided (B: element(k,n) at base + n + k*kstride): raw fetch.
static __device__ __forceinline__ void
fetch_bfrag_f32_kstrided(const float* __restrict__ base, long kstride, int k0, int n0,
                         float x[16]) {
  const int lane = threadIdx.x & 31;
  const int l16 = lane & 15, hi = lane >> 4;
  const float* colp = base + (n0 + l16);
  const int g0 = k0 + hi * 8, g1 = g0 + 16;
#pragma unroll
  for (int t = 0; t < 8; ++t) {
    x[t]     = colp[(long)(g0 + t) * kstride];
    x[t + 8] = colp[(long)(g1 + t) * kstride];
  }
}

static __device__ __forceinline__ v16bf cvt16arr(const float x[16]) {
  v16bf f;
#pragma unroll
  for (int t = 0; t < 8; ++t) {
    v2bf p = pk_bf16(x[2 * t], x[2 * t + 1]);
    f[2 * t] = p[0]; f[2 * t + 1] = p[1];
  }
  return f;
}

// bf16 source, K contiguous: one 16B load per 8-element K-group, no conversion.
static __device__ __forceinline__ v16bf
load_frag_bf16_kcontig(const __bf16* __restrict__ base, long ostride, int o0, int k0) {
  const int lane = threadIdx.x & 31;
  const int l16 = lane & 15, hi = lane >> 4;
  const __bf16* rowp = base + (long)(o0 + l16) * ostride;
  const int g0 = k0 + hi * 8, g1 = g0 + 16;
  union { float4 v; __bf16 h[8]; } u0, u1;
  u0.v = *(const float4*)(rowp + g0);
  u1.v = *(const float4*)(rowp + g1);
  v16bf f;
#pragma unroll
  for (int t = 0; t < 8; ++t) { f[t] = u0.h[t]; f[t + 8] = u1.h[t]; }
  return f;
}

// ---------------------------------------------------------------------------
// Streaming f32 -> bf16 (8 elems/thread, coalesced b128 in/out).
__global__ __launch_bounds__(256)
void f32_to_bf16(const float* __restrict__ in, __bf16* __restrict__ out, long n) {
  const long i = ((long)blockIdx.x * 256 + threadIdx.x) * 8;
  if (i + 8 <= n) {
    float4 a = *(const float4*)(in + i);
    float4 b = *(const float4*)(in + i + 4);
    union { v2bf p[4]; float4 v; } u;
    u.p[0] = pk_bf16(a.x, a.y);
    u.p[1] = pk_bf16(a.z, a.w);
    u.p[2] = pk_bf16(b.x, b.y);
    u.p[3] = pk_bf16(b.z, b.w);
    *(float4*)(out + i) = u.v;
  }
}

// out[m,n] = sum_k X[m,k]*W[n,k] + bias[n].  X: MxK bf16, W: NxK bf16.
// mode 0: f32 row-major MxN. mode 1: bf16 scattered to [B,NH,S,DH].
// mode 2: bf16 scattered transposed to [B,NH,DH,S] (for V).
__global__ __launch_bounds__(256)
void gemm_xwt(const __bf16* __restrict__ X, const __bf16* __restrict__ W,
              const float* __restrict__ bias, float* __restrict__ outf,
              __bf16* __restrict__ outh, int N, int K, int mode) {
  const int wave = threadIdx.x >> 5;
  const int lane = threadIdx.x & 31;
  const int l16 = lane & 15, hi = lane >> 4;
  const int m0 = blockIdx.x * 64 + (wave & 3) * 16;
  const int n0 = blockIdx.y * 64 + (wave >> 2) * 32;
  v8f acc0 = {}, acc1 = {};
  v16bf a  = load_frag_bf16_kcontig(X, K, m0, 0);
  v16bf b0 = load_frag_bf16_kcontig(W, K, n0, 0);
  v16bf b1 = load_frag_bf16_kcontig(W, K, n0 + 16, 0);
  for (int kk = 0; kk < K - 32; kk += 32) {      // peeled pipeline, no in-loop branch
    v16bf na  = load_frag_bf16_kcontig(X, K, m0, kk + 32);
    v16bf nb0 = load_frag_bf16_kcontig(W, K, n0, kk + 32);
    v16bf nb1 = load_frag_bf16_kcontig(W, K, n0 + 16, kk + 32);
    acc0 = wmma_bf16(a, b0, acc0);
    acc1 = wmma_bf16(a, b1, acc1);
    a = na; b0 = nb0; b1 = nb1;
  }
  acc0 = wmma_bf16(a, b0, acc0);
  acc1 = wmma_bf16(a, b1, acc1);
#pragma unroll
  for (int r = 0; r < 8; ++r) {
    const int row = m0 + r + hi * 8;
    const int c0 = n0 + l16, c1 = n0 + 16 + l16;
    const float v0 = acc0[r] + bias[c0];
    const float v1 = acc1[r] + bias[c1];
    if (mode == 1) {
      const int bb = row >> 9, s = row & (S_ - 1);
      outh[(((long)bb * NH_ + (c0 >> 6)) * S_ + s) * DH_ + (c0 & 63)] = f2bf(v0);
      outh[(((long)bb * NH_ + (c1 >> 6)) * S_ + s) * DH_ + (c1 & 63)] = f2bf(v1);
    } else if (mode == 2) {
      const int bb = row >> 9, s = row & (S_ - 1);
      outh[(((long)bb * NH_ + (c0 >> 6)) * DH_ + (c0 & 63)) * S_ + s] = f2bf(v0);
      outh[(((long)bb * NH_ + (c1 >> 6)) * DH_ + (c1 & 63)) * S_ + s] = f2bf(v1);
    } else {
      outf[(long)row * N + c0] = v0;
      outf[(long)row * N + c1] = v1;
    }
  }
}

// qs[b,h,i,d] = q[b,h,i,d] * Kp[h,d]   (bf16 in, bf16 out)
__global__ __launch_bounds__(256)
void qs_prep(const __bf16* __restrict__ q, const float* __restrict__ Kp,
             __bf16* __restrict__ qs) {
  const long idx = (long)blockIdx.x * 256 + threadIdx.x;
  const int d = (int)(idx & 63);
  const int h = (int)((idx >> 15) & 15);
  qs[idx] = f2bf(bf2f(q[idx]) * Kp[h * DH_ + d]);
}

// scores[b,h,i,j] = sum_d q[b,h,i,d]*k[b,h,j,d]   per (b,h): 512x512x64
__global__ __launch_bounds__(256)
void scores_base(const __bf16* __restrict__ q, const __bf16* __restrict__ k,
                 float* __restrict__ sc) {
  const int bh = blockIdx.z;
  const int wave = threadIdx.x >> 5;
  const int lane = threadIdx.x & 31;
  const int l16 = lane & 15, hi = lane >> 4;
  const __bf16* qh = q + (long)bh * S_ * DH_;
  const __bf16* kh = k + (long)bh * S_ * DH_;
  float* sh = sc + (long)bh * S_ * S_;
  const int m0 = blockIdx.x * 64 + (wave & 3) * 16;   // i
  const int n0 = blockIdx.y * 64 + (wave >> 2) * 32;  // j
  v8f acc0 = {}, acc1 = {};
#pragma unroll
  for (int kk = 0; kk < DH_; kk += 32) {
    v16bf a  = load_frag_bf16_kcontig(qh, DH_, m0, kk);
    v16bf b0 = load_frag_bf16_kcontig(kh, DH_, n0, kk);       // B[d,j]=K[j,d]
    v16bf b1 = load_frag_bf16_kcontig(kh, DH_, n0 + 16, kk);
    acc0 = wmma_bf16(a, b0, acc0);
    acc1 = wmma_bf16(a, b1, acc1);
  }
#pragma unroll
  for (int r = 0; r < 8; ++r) {
    const int row = m0 + r + hi * 8;
    sh[(long)row * S_ + n0 + l16]      = acc0[r];
    sh[(long)row * S_ + n0 + 16 + l16] = acc1[r];
  }
}

// scores[b,h,i,j] += sum_d qs[b,h,i,d]*rel[b,j,i,d]
// Per (b,i): M=heads(16), N=j(512), K=d(64). One WG per (b,i).
__global__ __launch_bounds__(256)
void scores_rel(const __bf16* __restrict__ qs, const float* __restrict__ rel,
                float* __restrict__ sc) {
  const int b = blockIdx.x >> 9;
  const int i = blockIdx.x & (S_ - 1);
  const int wave = threadIdx.x >> 5;
  const int lane = threadIdx.x & 31;
  const int l16 = lane & 15, hi = lane >> 4;
  const __bf16* qa = qs  + (long)b * NH_ * S_ * DH_ + (long)i * DH_;  // (h,d)
  const float*  rb = rel + (long)b * S_ * S_ * DH_ + (long)i * DH_;   // (d,j): +j*S*DH+d
  float* sbi       = sc  + (long)b * NH_ * S_ * S_ + (long)i * S_;    // (h,j)
  const int n0 = wave * 64;
  v8f acc[4] = {};
#pragma unroll
  for (int kk = 0; kk < DH_; kk += 32) {
    v16bf a = load_frag_bf16_kcontig(qa, (long)S_ * DH_, 0, kk);
    // fetch all raw f32 payloads first so the 16 b128 loads batch in flight
    f4x4 r0 = fetch_bfrag_f32_kcontig(rb, (long)S_ * DH_, kk, n0);
    f4x4 r1 = fetch_bfrag_f32_kcontig(rb, (long)S_ * DH_, kk, n0 + 16);
    f4x4 r2 = fetch_bfrag_f32_kcontig(rb, (long)S_ * DH_, kk, n0 + 32);
    f4x4 r3 = fetch_bfrag_f32_kcontig(rb, (long)S_ * DH_, kk, n0 + 48);
    acc[0] = wmma_bf16(a, cvt16(r0), acc[0]);
    acc[1] = wmma_bf16(a, cvt16(r1), acc[1]);
    acc[2] = wmma_bf16(a, cvt16(r2), acc[2]);
    acc[3] = wmma_bf16(a, cvt16(r3), acc[3]);
  }
#pragma unroll
  for (int t = 0; t < 4; ++t) {
#pragma unroll
    for (int r = 0; r < 8; ++r) {
      const int h = r + hi * 8;
      const int j = n0 + t * 16 + l16;
      sbi[(long)h * S_ * S_ + j] += acc[t][r];
    }
  }
}

// Row softmax over j (512), scale 1/sqrt(64); reads f32 scores, writes bf16 P.
__global__ __launch_bounds__(256)
void softmax_rows(const float* __restrict__ sc, __bf16* __restrict__ p) {
  __shared__ float red[256];
  const float* row = sc + (long)blockIdx.x * S_;
  __bf16* prow = p + (long)blockIdx.x * S_;
  const int t = threadIdx.x;
  const float x0 = row[t] * 0.125f;
  const float x1 = row[t + 256] * 0.125f;
  red[t] = fmaxf(x0, x1);
  __syncthreads();
  for (int s = 128; s > 0; s >>= 1) {
    if (t < s) red[t] = fmaxf(red[t], red[t + s]);
    __syncthreads();
  }
  const float m = red[0];
  __syncthreads();
  const float e0 = __expf(x0 - m), e1 = __expf(x1 - m);
  red[t] = e0 + e1;
  __syncthreads();
  for (int s = 128; s > 0; s >>= 1) {
    if (t < s) red[t] += red[t + s];
    __syncthreads();
  }
  const float inv = 1.0f / red[0];
  prow[t]       = f2bf(e0 * inv);
  prow[t + 256] = f2bf(e1 * inv);
}

// ctx[b,i,h*64+d] = sum_j P[b,h,i,j]*vT[b,h,d,j]   per (b,h): 512x64x512
// vT transposed layout makes the B operand contiguous b128 loads.
__global__ __launch_bounds__(256)
void out_pv(const __bf16* __restrict__ p, const __bf16* __restrict__ vt,
            float* __restrict__ ctx) {
  const int bh = blockIdx.y;
  const int wave = threadIdx.x >> 5;
  const int lane = threadIdx.x & 31;
  const int l16 = lane & 15, hi = lane >> 4;
  const __bf16* P   = p  + (long)bh * S_ * S_;
  const __bf16* vth = vt + (long)bh * DH_ * S_;
  const int m0 = blockIdx.x * 64 + (wave & 3) * 16;  // i
  const int n0 = (wave >> 2) * 32;                   // d
  v8f acc0 = {}, acc1 = {};
  v16bf a  = load_frag_bf16_kcontig(P, S_, m0, 0);
  v16bf b0 = load_frag_bf16_kcontig(vth, S_, n0, 0);        // B[j,d]=vT[d,j]
  v16bf b1 = load_frag_bf16_kcontig(vth, S_, n0 + 16, 0);
  for (int kk = 0; kk < S_ - 32; kk += 32) {
    v16bf na  = load_frag_bf16_kcontig(P, S_, m0, kk + 32);
    v16bf nb0 = load_frag_bf16_kcontig(vth, S_, n0, kk + 32);
    v16bf nb1 = load_frag_bf16_kcontig(vth, S_, n0 + 16, kk + 32);
    acc0 = wmma_bf16(a, b0, acc0);
    acc1 = wmma_bf16(a, b1, acc1);
    a = na; b0 = nb0; b1 = nb1;
  }
  acc0 = wmma_bf16(a, b0, acc0);
  acc1 = wmma_bf16(a, b1, acc1);
  const int b = bh >> 4, h = bh & 15;
#pragma unroll
  for (int r = 0; r < 8; ++r) {
    const int i = m0 + r + hi * 8;
    float* o = ctx + ((long)b * S_ + i) * HID_ + h * DH_;
    o[n0 + l16]      = acc0[r];
    o[n0 + 16 + l16] = acc1[r];
  }
}

// ctx[b,i,h*64+d] += Kp[h,d] * sum_j P[b,h,i,j]*rel[b,j,i,d]
// Per (b,i): M=heads(16), N=d(64), K=j(512). 4 waves, one 16-wide n-tile each.
__global__ __launch_bounds__(128)
void out_rel(const __bf16* __restrict__ p, const float* __restrict__ rel,
             const float* __restrict__ Kp, float* __restrict__ ctx) {
  const int b = blockIdx.x >> 9;
  const int i = blockIdx.x & (S_ - 1);
  const int wave = threadIdx.x >> 5;
  const int lane = threadIdx.x & 31;
  const int l16 = lane & 15, hi = lane >> 4;
  const __bf16* Pa = p   + (long)b * NH_ * S_ * S_ + (long)i * S_;   // (h,j)
  const float*  rb = rel + (long)b * S_ * S_ * DH_ + (long)i * DH_;  // (j,d): +j*S*DH+d
  const int n0 = wave * 16;
  v8f acc = {};
  float x[16];
  fetch_bfrag_f32_kstrided(rb, (long)S_ * DH_, 0, n0, x);
  v16bf a = load_frag_bf16_kcontig(Pa, (long)S_ * S_, 0, 0);
  for (int kk = 0; kk < S_ - 32; kk += 32) {
    float nx[16];
    fetch_bfrag_f32_kstrided(rb, (long)S_ * DH_, kk + 32, n0, nx);
    v16bf na = load_frag_bf16_kcontig(Pa, (long)S_ * S_, 0, kk + 32);
    acc = wmma_bf16(a, cvt16arr(x), acc);
#pragma unroll
    for (int t = 0; t < 16; ++t) x[t] = nx[t];
    a = na;
  }
  acc = wmma_bf16(a, cvt16arr(x), acc);
#pragma unroll
  for (int r = 0; r < 8; ++r) {
    const int h = r + hi * 8;
    const int d = n0 + l16;
    ctx[((long)b * S_ + i) * HID_ + h * DH_ + d] += acc[r] * Kp[h * DH_ + d];
  }
}

extern "C" void kernel_launch(void* const* d_in, const int* in_sizes, int n_in,
                              void* d_out, int out_size, void* d_ws, size_t ws_size,
                              hipStream_t stream) {
  (void)in_sizes; (void)n_in; (void)out_size; (void)ws_size;
  const float* key_   = (const float*)d_in[0];
  const float* value_ = (const float*)d_in[1];
  const float* query_ = (const float*)d_in[2];
  const float* rel    = (const float*)d_in[3];
  // d_in[4] = dropout (== 0) -> ignored
  const float* Wq = (const float*)d_in[5];
  const float* bq = (const float*)d_in[6];
  const float* Wk = (const float*)d_in[7];
  const float* bk = (const float*)d_in[8];
  const float* Wv = (const float*)d_in[9];
  const float* bv = (const float*)d_in[10];
  const float* Kp = (const float*)d_in[11];
  const float* Wo = (const float*)d_in[12];
  const float* bo = (const float*)d_in[13];
  float* out = (float*)d_out;

  const long HW  = (long)HID_ * HID_;             // 1M elems (weights)
  const long XE  = (long)B_ * S_ * HID_;          // 2M elems (activations)
  const long QKV = (long)B_ * NH_ * S_ * DH_;     // 2M elems
  const long SCN = (long)B_ * NH_ * S_ * S_;      // 16M elems

  __bf16* wqb = (__bf16*)d_ws;
  __bf16* wkb = wqb + HW;
  __bf16* wvb = wkb + HW;
  __bf16* wob = wvb + HW;
  __bf16* xqb = wob + HW;
  __bf16* xkb = xqb + XE;
  __bf16* xvb = xkb + XE;
  __bf16* qb  = xvb + XE;
  __bf16* kb  = qb + QKV;
  __bf16* vtb = kb + QKV;                          // V transposed [B,NH,DH,S]
  __bf16* qsb = vtb + QKV;
  __bf16* pbf = qsb + QKV;                         // 32 MiB
  __bf16* ctxb = pbf + SCN;
  float*  sc  = (float*)(ctxb + XE);               // 64 MiB
  float*  ctx = sc + SCN;                          // 8 MiB

  const dim3 blk(256);
  const dim3 gGemm(2048 / 64, 1024 / 64);
  const unsigned gW = (unsigned)(HW / (8 * 256));  // 512 blocks
  const unsigned gX = (unsigned)(XE / (8 * 256));  // 1024 blocks

  hipLaunchKernelGGL(f32_to_bf16, dim3(gW), blk, 0, stream, Wq, wqb, HW);
  hipLaunchKernelGGL(f32_to_bf16, dim3(gW), blk, 0, stream, Wk, wkb, HW);
  hipLaunchKernelGGL(f32_to_bf16, dim3(gW), blk, 0, stream, Wv, wvb, HW);
  hipLaunchKernelGGL(f32_to_bf16, dim3(gW), blk, 0, stream, Wo, wob, HW);
  hipLaunchKernelGGL(f32_to_bf16, dim3(gX), blk, 0, stream, query_, xqb, XE);
  hipLaunchKernelGGL(f32_to_bf16, dim3(gX), blk, 0, stream, key_,   xkb, XE);
  hipLaunchKernelGGL(f32_to_bf16, dim3(gX), blk, 0, stream, value_, xvb, XE);

  hipLaunchKernelGGL(gemm_xwt, gGemm, blk, 0, stream, xqb, wqb, bq, nullptr, qb, 1024, 1024, 1);
  hipLaunchKernelGGL(gemm_xwt, gGemm, blk, 0, stream, xkb, wkb, bk, nullptr, kb, 1024, 1024, 1);
  hipLaunchKernelGGL(gemm_xwt, gGemm, blk, 0, stream, xvb, wvb, bv, nullptr, vtb, 1024, 1024, 2);
  hipLaunchKernelGGL(qs_prep, dim3((unsigned)(QKV / 256)), blk, 0, stream, qb, Kp, qsb);
  hipLaunchKernelGGL(scores_base, dim3(8, 8, B_ * NH_), blk, 0, stream, qb, kb, sc);
  hipLaunchKernelGGL(scores_rel, dim3(B_ * S_), blk, 0, stream, qsb, rel, sc);
  hipLaunchKernelGGL(softmax_rows, dim3(B_ * NH_ * S_), blk, 0, stream, sc, pbf);
  hipLaunchKernelGGL(out_pv, dim3(8, B_ * NH_), blk, 0, stream, pbf, vtb, ctx);
  hipLaunchKernelGGL(out_rel, dim3(B_ * S_), dim3(128), 0, stream, pbf, rel, Kp, ctx);
  hipLaunchKernelGGL(f32_to_bf16, dim3(gX), blk, 0, stream, ctx, ctxb, XE);
  hipLaunchKernelGGL(gemm_xwt, gGemm, blk, 0, stream, ctxb, wob, bo, out, nullptr, 1024, 1024, 0);
}
